// Model_18511309046127
// MI455X (gfx1250) — matrix operand order repaired and checked by reading, not compiled, not run on hardware
//
#include <hip/hip_runtime.h>
#include <hip/hip_bf16.h>

// ---------------------------------------------------------------------------
// GCN forward (2x GCNConv + mean-pool + MLP) for MI455X (gfx1250, wave32).
// GEMMs: V_WMMA_F32_16X16X32_F16, with A and B pre-packed into the ISA 7.12.2
// fragment layouts so every operand fetch is a contiguous 32B load (b128 x2).
// Each wave holds all 4 B fragments in registers and sweeps 4 M-tiles.
// M-tiles are always full (M % 16 == 0), so D stores are unguarded b32 stores
// at immediate offsets from one base address.
// Edge aggregation: fp32 HW atomics (working set 25.6MB << 192MB L2).
// ---------------------------------------------------------------------------

typedef __attribute__((ext_vector_type(16))) _Float16 v16h;
typedef __attribute__((ext_vector_type(8)))  float    v8f;

typedef long long i64;   // reference edge_index / batch are int64

#define DIM 128
#define NGRAPH 128

// A-fragment column permutation (16-bit A 16x32, ISA 7.12.2):
// lane half h, element e<8  -> K = ks*32 + h*8 + e
// lane half h, element e>=8 -> K = ks*32 + 16 + h*8 + (e-8)
// Stored so lane's v16h is contiguous: col = ks*32 + h*16 + e.
__device__ __forceinline__ int afrag_col(int k) {
    int ks = k >> 5, r = k & 31;
    int group  = r >> 4;          // 0: K%32 in [0,16), 1: [16,32)
    int within = r & 15;
    int half   = within >> 3;
    int e      = (within & 7) | (group << 3);
    return ks * 32 + half * 16 + e;
}

// ---------------- elementwise helpers ----------------

__global__ __launch_bounds__(256) void k_zero_f32(float* p, long long n) {
    long long t = (long long)blockIdx.x * blockDim.x + threadIdx.x;
    if (t < n) p[t] = 0.0f;
}

// f32 -> f16 with A-fragment permutation of the 128-wide rows
__global__ __launch_bounds__(256) void k_cvt_afrag(const float* __restrict__ in,
                                                   _Float16* __restrict__ out, long long n) {
    long long t = (long long)blockIdx.x * blockDim.x + threadIdx.x;
    if (t < n) {
        long long row = t >> 7;
        int k = (int)(t & (DIM - 1));
        out[row * DIM + afrag_col(k)] = (_Float16)in[t];
    }
}

// Pack B[128x128] (f32 row-major, K-major) into per-lane-contiguous WMMA
// fragments, fused with the f16 convert:
//   Bp[((nt*4 + ks)*32 + lane)*16 + e] = W[(ks*32 + (e&8)*2 + (lane>>4)*8 + (e&7))*128 + nt*16 + (lane&15)]
__global__ __launch_bounds__(256) void k_pack_B(const float* __restrict__ W,
                                                _Float16* __restrict__ Bp) {
    int t = blockIdx.x * blockDim.x + threadIdx.x;   // 16384 threads
    if (t >= DIM * DIM) return;
    int e    = t & 15;
    int lane = (t >> 4) & 31;
    int ks   = (t >> 9) & 3;
    int nt   = t >> 11;
    int k = ks * 32 + (e & 8) * 2 + (lane >> 4) * 8 + (e & 7);
    int n = nt * 16 + (lane & 15);
    Bp[t] = (_Float16)W[k * DIM + n];
}

// degree -> dinv = rsqrt(deg + 1)  (+1 = self-loop weight, so deg > 0 always)
__global__ __launch_bounds__(256) void k_deg_accum(const i64* __restrict__ dst,
                                                   const float* __restrict__ ea,
                                                   float* __restrict__ deg, long long E) {
    long long t = (long long)blockIdx.x * blockDim.x + threadIdx.x;
    if (t < E) unsafeAtomicAdd(&deg[dst[t]], ea[t]);
}

__global__ __launch_bounds__(256) void k_deg_finalize(float* __restrict__ dinv, int N) {
    int t = blockIdx.x * blockDim.x + threadIdx.x;
    if (t < N) dinv[t] = rsqrtf(dinv[t] + 1.0f);
}

// ---------------- WMMA GEMM: C[M x 128] = A[M x 128] * B[128 x 128] ----------
// A is f16 in A-fragment-permuted row layout; Bp is the packed fragment array.
// Block = 8 waves; wave = N-tile; block covers 4 consecutive M-tiles.
// Requires M % 16 == 0 (true here: 50000, 128) -> executed tiles are full,
// stores need no per-row guards.

__global__ __launch_bounds__(256) void k_wmma_gemm(const _Float16* __restrict__ A,
                                                   const _Float16* __restrict__ Bp,
                                                   float* __restrict__ C, int M) {
    const int lane = threadIdx.x & 31;
    const int nt   = threadIdx.x >> 5;   // wave id == N tile (0..7)
    const int half = lane >> 4;
    const int l15  = lane & 15;

    // all 4 B fragments for this N-tile, kept in VGPRs across the M sweep
    v16h bfrag[4];
    #pragma unroll
    for (int ks = 0; ks < 4; ++ks)
        bfrag[ks] = *(const v16h*)(Bp + (((nt * 4 + ks) * 32 + lane) << 4));

    const int n = nt * 16 + l15;
    #pragma unroll
    for (int sub = 0; sub < 4; ++sub) {
        const int m0 = (blockIdx.x * 4 + sub) * 16;
        if (m0 >= M) break;                       // uniform tile-level guard
        const _Float16* arow = A + (size_t)(m0 + l15) * DIM + half * 16;

        v8f acc = {};
        #pragma unroll
        for (int ks = 0; ks < 4; ++ks) {
            v16h a = *(const v16h*)(arow + ks * 32);   // contiguous 32B fragment
            acc = __builtin_amdgcn_wmma_f32_16x16x32_f16(
                false, a, false, bfrag[ks], (short)0, acc, false, false);
        }

        // D (16x16 f32): VGPR r -> row = half*8 + r, col = l15.
        // Full tile: one base address, 8 stores at immediate offsets (r*DIM).
        float* crow = C + (size_t)(m0 + half * 8) * DIM + n;
        #pragma unroll
        for (int r = 0; r < 8; ++r) crow[r * DIM] = acc[r];
    }
}

// ---------------- GCN aggregation ----------------

// Self-loop term: out[n][d] = dinv[n]^2 * h[n][d]
__global__ __launch_bounds__(256) void k_agg_init(const float* __restrict__ H,
                                                  const float* __restrict__ dinv,
                                                  float* __restrict__ out, long long n) {
    long long t = (long long)blockIdx.x * blockDim.x + threadIdx.x;
    if (t < n) {
        float di = dinv[t >> 7];
        out[t] = di * di * H[t];
    }
}

// Edge scatter: out[dst] += (dinv[src]*ea*dinv[dst]) * h[src].
// One wave per edge; each lane owns a float4 chunk -> the wave's gather and
// atomics each touch one contiguous 512B row (coalesced, L2-resident).
__global__ __launch_bounds__(256) void k_scatter(const i64* __restrict__ src,
                                                 const i64* __restrict__ dst,
                                                 const float* __restrict__ ea,
                                                 const float* __restrict__ dinv,
                                                 const float* __restrict__ H,
                                                 float* __restrict__ out, long long E) {
    long long t = (long long)blockIdx.x * blockDim.x + threadIdx.x;
    if (t >= E * 32) return;
    long long e = t >> 5;
    int l = (int)(t & 31);
    long long s = src[e], d = dst[e];
    float coef = dinv[s] * ea[e] * dinv[d];
    const float4 hv = *(const float4*)(H + s * DIM + l * 4);
    float* o = out + d * DIM + l * 4;
    unsafeAtomicAdd(o + 0, coef * hv.x);
    unsafeAtomicAdd(o + 1, coef * hv.y);
    unsafeAtomicAdd(o + 2, coef * hv.z);
    unsafeAtomicAdd(o + 3, coef * hv.w);
}

// out = relu(in + bias) -> optional f32 copy and/or A-fragment-permuted f16
__global__ __launch_bounds__(256) void k_bias_relu(const float* __restrict__ in,
                                                   const float* __restrict__ bias,
                                                   float* __restrict__ out32,
                                                   _Float16* __restrict__ out16,
                                                   long long n) {
    long long t = (long long)blockIdx.x * blockDim.x + threadIdx.x;
    if (t < n) {
        int k = (int)(t & (DIM - 1));
        float v = in[t] + bias[k];
        v = v > 0.0f ? v : 0.0f;
        if (out32) out32[t] = v;
        if (out16) out16[(t & ~(long long)(DIM - 1)) + afrag_col(k)] = (_Float16)v;
    }
}

// ---------------- pooling + classifier ----------------

__global__ __launch_bounds__(256) void k_pool_accum(const float* __restrict__ H,
                                                    const i64* __restrict__ batch,
                                                    float* __restrict__ gsum,
                                                    float* __restrict__ gcnt, long long n) {
    long long t = (long long)blockIdx.x * blockDim.x + threadIdx.x;
    if (t < n) {
        long long node = t >> 7;
        int dim = (int)(t & (DIM - 1));
        long long g = batch[node];
        unsafeAtomicAdd(&gsum[g * DIM + dim], H[t]);
        if (dim == 0) unsafeAtomicAdd(&gcnt[g], 1.0f);
    }
}

// mean + f16 convert, written in A-fragment order for the classifier GEMM
__global__ __launch_bounds__(256) void k_pool_div(const float* __restrict__ gsum,
                                                  const float* __restrict__ gcnt,
                                                  _Float16* __restrict__ Gh, int n) {
    int t = blockIdx.x * blockDim.x + threadIdx.x;
    if (t < n) {
        int g = t >> 7, k = t & (DIM - 1);
        float c = gcnt[g];
        Gh[g * DIM + afrag_col(k)] = (_Float16)(gsum[t] / fmaxf(c, 1.0f));
    }
}

// final tiny FC: out[g][o] = sum_d G1[g][d] * Wc2[d][o] + bc2[o]   (128x4)
__global__ __launch_bounds__(256) void k_final_fc(const float* __restrict__ G1,
                                                  const float* __restrict__ Wc2,
                                                  const float* __restrict__ bc2,
                                                  float* __restrict__ out) {
    int t = blockIdx.x * blockDim.x + threadIdx.x;
    if (t >= NGRAPH * 4) return;
    int g = t >> 2, o = t & 3;
    float s = bc2[o];
    #pragma unroll 8
    for (int d = 0; d < DIM; ++d) s += G1[g * DIM + d] * Wc2[d * 4 + o];
    out[t] = s;
}

// ---------------------------------------------------------------------------

extern "C" void kernel_launch(void* const* d_in, const int* in_sizes, int n_in,
                              void* d_out, int out_size, void* d_ws, size_t ws_size,
                              hipStream_t stream) {
    // inputs (setup_inputs order)
    const float* x   = (const float*)d_in[0];
    const i64*  eidx = (const i64*) d_in[1];          // [2, E] int64
    const float* ea  = (const float*)d_in[2];          // edge_attr (used as weight)
    // d_in[3] edge_weight: unused by reference
    const i64*  batch= (const i64*) d_in[4];
    const float* W0  = (const float*)d_in[5];
    const float* b0  = (const float*)d_in[6];
    const float* W1  = (const float*)d_in[7];
    const float* b1  = (const float*)d_in[8];
    const float* Wc1 = (const float*)d_in[9];
    const float* bc1 = (const float*)d_in[10];
    const float* Wc2 = (const float*)d_in[11];
    const float* bc2 = (const float*)d_in[12];
    float* out = (float*)d_out;

    const long long N = in_sizes[0] / DIM;     // 50000 (multiple of 16)
    const long long E = in_sizes[2];           // 800000
    const long long ND = N * DIM;

    // workspace carve-out (~64.6 MB)
    char* p = (char*)d_ws;
    auto alloc = [&](size_t bytes) { char* r = p; p += (bytes + 255) & ~(size_t)255; return (void*)r; };
    _Float16* Xh   = (_Float16*)alloc(ND * sizeof(_Float16));       // f16 GEMM input (A-frag layout)
    float*    Hbuf = (float*)   alloc(ND * sizeof(float));          // GEMM output
    float*    Abuf = (float*)   alloc(ND * sizeof(float));          // aggregation
    float*    dinv = (float*)   alloc(N * sizeof(float));           // deg -> dinv
    _Float16* B0p  = (_Float16*)alloc(DIM * DIM * sizeof(_Float16));// packed B fragments
    _Float16* B1p  = (_Float16*)alloc(DIM * DIM * sizeof(_Float16));
    _Float16* Bc1p = (_Float16*)alloc(DIM * DIM * sizeof(_Float16));
    float*    gsum = (float*)   alloc(NGRAPH * DIM * sizeof(float));
    float*    gcnt = (float*)   alloc(NGRAPH * sizeof(float));
    _Float16* Gh   = (_Float16*)alloc(NGRAPH * DIM * sizeof(_Float16));
    float*    G1   = (float*)   alloc(NGRAPH * DIM * sizeof(float));

    const i64* esrc = eidx;
    const i64* edst = eidx + E;

    auto blk = [](long long n) { return (unsigned)((n + 255) / 256); };
    const unsigned gemmN   = (unsigned)((N / 16 + 3) / 4);   // 4 M-tiles per block
    const unsigned gemmCls = (NGRAPH / 16 + 3) / 4;

    // 1. symmetric normalization coefficients (shared by both layers)
    k_zero_f32    <<<blk(N), 256, 0, stream>>>(dinv, N);
    k_deg_accum   <<<blk(E), 256, 0, stream>>>(edst, ea, dinv, E);
    k_deg_finalize<<<blk(N), 256, 0, stream>>>(dinv, (int)N);

    // 2. pack weights into WMMA B-fragment layout (fused f32->f16)
    k_pack_B<<<blk(DIM * DIM), 256, 0, stream>>>(W0,  B0p);
    k_pack_B<<<blk(DIM * DIM), 256, 0, stream>>>(W1,  B1p);
    k_pack_B<<<blk(DIM * DIM), 256, 0, stream>>>(Wc1, Bc1p);

    // 3. layer 0: H = X @ W0 (WMMA), aggregate, bias+relu -> f16 for layer 1
    k_cvt_afrag<<<blk(ND), 256, 0, stream>>>(x, Xh, ND);
    k_wmma_gemm<<<gemmN, 256, 0, stream>>>(Xh, B0p, Hbuf, (int)N);
    k_agg_init <<<blk(ND), 256, 0, stream>>>(Hbuf, dinv, Abuf, ND);
    k_scatter  <<<blk(E * 32), 256, 0, stream>>>(esrc, edst, ea, dinv, Hbuf, Abuf, E);
    k_bias_relu<<<blk(ND), 256, 0, stream>>>(Abuf, b0, nullptr, Xh, ND);

    // 4. layer 1: H = relu0 @ W1 (WMMA), aggregate, bias+relu -> f32 for pooling
    k_wmma_gemm<<<gemmN, 256, 0, stream>>>(Xh, B1p, Hbuf, (int)N);
    k_agg_init <<<blk(ND), 256, 0, stream>>>(Hbuf, dinv, Abuf, ND);
    k_scatter  <<<blk(E * 32), 256, 0, stream>>>(esrc, edst, ea, dinv, Hbuf, Abuf, E);
    k_bias_relu<<<blk(ND), 256, 0, stream>>>(Abuf, b1, Hbuf, nullptr, ND);

    // 5. global mean pool  (gsum and gcnt contiguous -> one zero pass)
    k_zero_f32  <<<blk(NGRAPH * DIM + NGRAPH), 256, 0, stream>>>(gsum, NGRAPH * DIM + NGRAPH);
    k_pool_accum<<<blk(ND), 256, 0, stream>>>(Hbuf, batch, gsum, gcnt, ND);
    k_pool_div  <<<blk(NGRAPH * DIM), 256, 0, stream>>>(gsum, gcnt, Gh, NGRAPH * DIM);

    // 6. classifier: relu(G @ Wc1 + bc1) @ Wc2 + bc2
    k_wmma_gemm<<<gemmCls, 256, 0, stream>>>(Gh, Bc1p, G1, NGRAPH);
    k_bias_relu<<<blk(NGRAPH * DIM), 256, 0, stream>>>(G1, bc1, G1, nullptr, NGRAPH * DIM);
    k_final_fc <<<blk(NGRAPH * 4), 256, 0, stream>>>(G1, Wc2, bc2, out);
}
